// CrossModalHyperedgeInteraction_9242769621319
// MI455X (gfx1250) — matrix operand order repaired and checked by reading, not compiled, not run on hardware
//
#include <hip/hip_runtime.h>
#include <hip/hip_bf16.h>

typedef __attribute__((ext_vector_type(16))) _Float16 v16h;
typedef __attribute__((ext_vector_type(8)))  _Float16 v8h;
typedef __attribute__((ext_vector_type(8)))  float    v8f;

#define D_MODEL 512
#define ATT_SCALE 0.04419417382415922f  // 1/sqrt(512)

// ---------------------------------------------------------------------------
// WMMA fragment loaders for V_WMMA_F32_16X16X32_F16, per CDNA5 ISA 7.12.2.
// A (16x32 f16): lane L holds row (L&15); halves [kb..kb+7] and [16+kb..23+kb],
//                kb = 8*(L>=16).
// B (32x16 f16): lane L holds column (L&15); halves [kb..kb+7] and [16+kb..23+kb],
//                kb = 8*(L>=16) (B tile stored K-contiguous per column).
// C/D (16x16 f32): VGPR r holds M = r + 8*(L>=16), N = L&15.
// ---------------------------------------------------------------------------
static __device__ __forceinline__ v16h frag_a(const _Float16* base, int ld) {
  const int lane = threadIdx.x & 31;
  const int row  = lane & 15;
  const int kb   = (lane >> 4) << 3;
  const _Float16* p = base + row * ld + kb;
  v8h lo = *(const v8h*)(p);
  v8h hi = *(const v8h*)(p + 16);
  v16h r;
#pragma unroll
  for (int i = 0; i < 8; ++i) { r[i] = lo[i]; r[i + 8] = hi[i]; }
  return r;
}

static __device__ __forceinline__ v16h frag_b(const _Float16* base, int ld) {
  const int lane = threadIdx.x & 31;
  const int col  = lane & 15;
  const int kb   = (lane >> 4) << 3;
  const _Float16* p = base + col * ld + kb;
  v8h lo = *(const v8h*)(p);
  v8h hi = *(const v8h*)(p + 16);
  v16h r;
#pragma unroll
  for (int i = 0; i < 8; ++i) { r[i] = lo[i]; r[i + 8] = hi[i]; }
  return r;
}

static __device__ __forceinline__ v8f wmma32(v16h a, v16h b, v8f c) {
  return __builtin_amdgcn_wmma_f32_16x16x32_f16(false, a, false, b, (short)0, c,
                                                false, false);
}

// ---------------------------------------------------------------------------
// CDNA5 async global->LDS copy (GLOBAL_LOAD_ASYNC_TO_LDS_B128, ASYNCcnt).
// Generic LDS pointers carry the wave-relative LDS byte offset in the low 32
// bits (ISA 10.2 aperture mapping), so the truncated pointer is the dsaddr.
// ---------------------------------------------------------------------------
static __device__ __forceinline__ void async_load_b128(_Float16* lds_dst,
                                                       const _Float16* gsrc) {
  unsigned ldsoff = (unsigned)(unsigned long long)lds_dst;
  asm volatile("global_load_async_to_lds_b128 %0, %1, off"
               :
               : "v"(ldsoff), "v"(gsrc)
               : "memory");
}

static __device__ __forceinline__ void wait_async0() {
  asm volatile("s_wait_asynccnt 0x0" ::: "memory");
}

// ---------------------------------------------------------------------------
// Projection GEMM: C[t,e] = sum_d E[t,d]*W[e,d] (+bias). blockIdx.y selects
// the weight (0=q,1=k,2=v,3=o). q/k/v written as f16 to workspace; o as f32
// into d_out (msg is accumulated on top later, relu applied last).
// Block tile 64(M tokens) x 64(N outputs); 8 waves as 4(M) x 2(N).
// Double-buffered LDS staging: next k-step's loads+cvt overlap current WMMAs.
// ---------------------------------------------------------------------------
__global__ __launch_bounds__(256) void proj_kernel(
    const float* __restrict__ E,
    const float* __restrict__ Wq, const float* __restrict__ Wk,
    const float* __restrict__ Wv, const float* __restrict__ Wo,
    const float* __restrict__ bq, const float* __restrict__ bk,
    const float* __restrict__ bv, const float* __restrict__ bo,
    _Float16* __restrict__ Qd, _Float16* __restrict__ Kd,
    _Float16* __restrict__ Vd, float* __restrict__ Po) {
  __shared__ _Float16 As[2][64 * 40];  // [token][d], padded stride 40
  __shared__ _Float16 Bs[2][64 * 40];  // [e][d],     padded stride 40

  const int t    = threadIdx.x;
  const int wv   = t >> 5;
  const int lane = t & 31;
  const int wm   = wv >> 1;   // 0..3 (M group)
  const int wn   = wv & 1;    // 0..1 (N group)
  const int m0   = (blockIdx.x >> 3) * 64;
  const int n0   = (blockIdx.x & 7) * 64;
  const int w    = blockIdx.y;

  const float* Wsel = (w == 0) ? Wq : (w == 1) ? Wk : (w == 2) ? Wv : Wo;
  const float* bsel = (w == 0) ? bq : (w == 1) ? bk : (w == 2) ? bv : bo;

  v8f acc0 = {0, 0, 0, 0, 0, 0, 0, 0};
  v8f acc1 = {0, 0, 0, 0, 0, 0, 0, 0};

  const int ar = t >> 2;         // staging row 0..63
  const int ac = (t & 3) << 3;   // staging col 0,8,16,24
  const float* esrc0 = E + (size_t)(m0 + ar) * D_MODEL + ac;
  const float* wsrc0 = Wsel + (size_t)(n0 + ar) * D_MODEL + ac;

  // prologue: stage k-step 0 into buffer 0
#pragma unroll
  for (int u = 0; u < 8; ++u) As[0][ar * 40 + ac + u] = (_Float16)esrc0[u];
#pragma unroll
  for (int u = 0; u < 8; ++u) Bs[0][ar * 40 + ac + u] = (_Float16)wsrc0[u];
  __syncthreads();

  for (int ks = 0; ks < 16; ++ks) {
    const int cur = ks & 1;
    const bool more = (ks < 15);
    float eN[8], wN[8];
    if (more) {
      const float* e = esrc0 + (ks + 1) * 32;
      const float* ww = wsrc0 + (ks + 1) * 32;
#pragma unroll
      for (int u = 0; u < 8; ++u) { eN[u] = e[u]; wN[u] = ww[u]; }
    }
    v16h a  = frag_a(&As[cur][wm * 16 * 40], 40);
    v16h b0 = frag_b(&Bs[cur][(wn * 32) * 40], 40);
    v16h b1 = frag_b(&Bs[cur][(wn * 32 + 16) * 40], 40);
    acc0 = wmma32(a, b0, acc0);
    acc1 = wmma32(a, b1, acc1);
    if (more) {
#pragma unroll
      for (int u = 0; u < 8; ++u) As[1 - cur][ar * 40 + ac + u] = (_Float16)eN[u];
#pragma unroll
      for (int u = 0; u < 8; ++u) Bs[1 - cur][ar * 40 + ac + u] = (_Float16)wN[u];
    }
    __syncthreads();
  }

  const int rowBase = m0 + wm * 16 + ((lane >> 4) << 3);
  const int colBase = n0 + wn * 32 + (lane & 15);
  auto store_tile = [&](v8f acc, int col) {
    const float bias = bsel[col];
#pragma unroll
    for (int r = 0; r < 8; ++r) {
      const size_t idx = (size_t)(rowBase + r) * D_MODEL + col;
      const float val  = acc[r] + bias;
      if (w == 0)      Qd[idx] = (_Float16)val;
      else if (w == 1) Kd[idx] = (_Float16)val;
      else if (w == 2) Vd[idx] = (_Float16)val;
      else             Po[idx] = val;
    }
  };
  store_tile(acc0, colBase);
  store_tile(acc1, colBase + 16);
}

// ---------------------------------------------------------------------------
// Cross attention for one (m,n) pair: block = (16 Q rows) x (full Kn, full D).
// Phase1: S = Q K^T in LDS (16 x Kn f32, Kn<=1024 -> <=64KB).
// Phase2: row softmax (scaled), P stored f16 in LDS.
// Phase3: O = P V, V chunks (32 tokens) transposed into LDS; wave w owns a
//         64-wide D slice. Result ACCUMULATED into out (proj_o already there).
// ---------------------------------------------------------------------------
__global__ __launch_bounds__(256) void attn_kernel(
    const _Float16* __restrict__ Q, const _Float16* __restrict__ K,
    const _Float16* __restrict__ V, float* __restrict__ out,
    int Km, int Kn, long qTokBase, long kTokBase, long outBase) {
  extern __shared__ char smem[];
  _Float16* Qs = (_Float16*)smem;                                   // 16*520
  float*    S  = (float*)(smem + 16 * 520 * 2);                     // 16*Kn
  _Float16* P  = (_Float16*)(smem + 16 * 520 * 2 + (size_t)16 * Kn * 4);
  _Float16* Vt = (_Float16*)((char*)P + (size_t)16 * (Kn + 8) * 2); // 512*40

  const int t    = threadIdx.x;
  const int wv   = t >> 5;
  const int lane = t & 31;
  const int qt   = blockIdx.x;
  const int b    = blockIdx.y;
  const long qRow0 = (long)b * Km + (long)qt * 16;
  const int Pld  = Kn + 8;

  // ---- stage Q tile (16 x 512 f16) via async global->LDS (no VGPR hop) ----
#pragma unroll
  for (int j = 0; j < 4; ++j) {
    const int linear = j * 2048 + t * 8;
    const int row = linear >> 9, d0 = linear & 511;
    async_load_b128(Qs + row * 520 + d0,
                    Q + (qTokBase + qRow0 + row) * D_MODEL + d0);
  }
  wait_async0();
  __syncthreads();

  // ---- phase 1: S = Q K^T (waves split Kn into 16-col subtiles) ----
  const int nsub = Kn >> 4;
  for (int st = wv; st < nsub; st += 8) {
    v8f acc = {0, 0, 0, 0, 0, 0, 0, 0};
    const _Float16* kbase = K + (kTokBase + (long)b * Kn + (long)st * 16) * D_MODEL;
    if (st + 8 < nsub)
      __builtin_prefetch((const void*)(kbase + (size_t)128 * D_MODEL), 0, 1);
#pragma unroll 4
    for (int ks = 0; ks < 16; ++ks) {
      v16h a  = frag_a(Qs + ks * 32, 520);
      v16h bb = frag_b(kbase + ks * 32, D_MODEL);
      acc = wmma32(a, bb, acc);
    }
    const int col = st * 16 + (lane & 15);
    const int rhi = (lane >> 4) << 3;
#pragma unroll
    for (int r = 0; r < 8; ++r) S[(size_t)(r + rhi) * Kn + col] = acc[r];
  }
  __syncthreads();

  // ---- phase 2: softmax(S * scale) per row; P = probs (f16) ----
  {
    const int row = t >> 4, l16 = t & 15;
    const float* srow = S + (size_t)row * Kn;
    _Float16*   prow = P + (size_t)row * Pld;
    float mx = -3.0e38f;
    for (int c = l16; c < Kn; c += 16) mx = fmaxf(mx, srow[c]);
#pragma unroll
    for (int m = 8; m >= 1; m >>= 1) mx = fmaxf(mx, __shfl_xor(mx, m, 16));
    float sum = 0.f;
    for (int c = l16; c < Kn; c += 16) {
      const float e = __expf((srow[c] - mx) * ATT_SCALE);
      sum += e;
      prow[c] = (_Float16)e;
    }
#pragma unroll
    for (int m = 8; m >= 1; m >>= 1) sum += __shfl_xor(sum, m, 16);
    const float inv = 1.0f / sum;
    for (int c = l16; c < Kn; c += 16)
      prow[c] = (_Float16)((float)prow[c] * inv);
  }

  // ---- phase 3: O = P V; wave wv owns D columns [wv*64, wv*64+64) ----
  v8f o0 = {0, 0, 0, 0, 0, 0, 0, 0};
  v8f o1 = {0, 0, 0, 0, 0, 0, 0, 0};
  v8f o2 = {0, 0, 0, 0, 0, 0, 0, 0};
  v8f o3 = {0, 0, 0, 0, 0, 0, 0, 0};
  const int nch = Kn >> 5;
  for (int kc = 0; kc < nch; ++kc) {
    __syncthreads();  // previous Vt reads done / P visible on first iter
#pragma unroll
    for (int j = 0; j < 8; ++j) {  // transpose 32-token V chunk into Vt[d][k]
      const int linear = j * 2048 + t * 8;
      const int kk = linear >> 9, d0 = linear & 511;
      v8h v = *(const v8h*)(V + (kTokBase + (long)b * Kn + (long)kc * 32 + kk) *
                                    D_MODEL + d0);
#pragma unroll
      for (int u = 0; u < 8; ++u) Vt[(d0 + u) * 40 + kk] = v[u];
    }
    __syncthreads();
    v16h a = frag_a(P + kc * 32, Pld);
    o0 = wmma32(a, frag_b(Vt + (wv * 64 + 0) * 40, 40), o0);
    o1 = wmma32(a, frag_b(Vt + (wv * 64 + 16) * 40, 40), o1);
    o2 = wmma32(a, frag_b(Vt + (wv * 64 + 32) * 40, 40), o2);
    o3 = wmma32(a, frag_b(Vt + (wv * 64 + 48) * 40, 40), o3);
  }

  // ---- epilogue: accumulate msg into out ----
  const int rhi = (lane >> 4) << 3;
  const int cl  = lane & 15;
  float* obase = out + outBase + qRow0 * D_MODEL;
  auto acc_store = [&](v8f o, int dc) {
#pragma unroll
    for (int r = 0; r < 8; ++r) {
      const size_t idx = (size_t)(r + rhi) * D_MODEL + wv * 64 + dc + cl;
      obase[idx] += o[r];
    }
  };
  acc_store(o0, 0);
  acc_store(o1, 16);
  acc_store(o2, 32);
  acc_store(o3, 48);
}

__global__ void relu_kernel(float* __restrict__ out, long n) {
  long i = (long)blockIdx.x * blockDim.x + threadIdx.x;
  const long stride = (long)gridDim.x * blockDim.x;
  for (; i < n; i += stride) out[i] = fmaxf(out[i], 0.0f);
}

// ---------------------------------------------------------------------------
extern "C" void kernel_launch(void* const* d_in, const int* in_sizes, int n_in,
                              void* d_out, int out_size, void* d_ws,
                              size_t ws_size, hipStream_t stream) {
  (void)in_sizes; (void)n_in; (void)out_size; (void)ws_size;
  const float* E[3] = {(const float*)d_in[0], (const float*)d_in[1],
                       (const float*)d_in[2]};
  const float* Wq = (const float*)d_in[3];
  const float* bq = (const float*)d_in[4];
  const float* Wk = (const float*)d_in[5];
  const float* bk = (const float*)d_in[6];
  const float* Wv = (const float*)d_in[7];
  const float* bv = (const float*)d_in[8];
  const float* Wo = (const float*)d_in[9];
  const float* bo = (const float*)d_in[10];

  static const int Kmods[3] = {1024, 768, 512};
  const long NT = 32L * (1024 + 768 + 512);  // 73728 tokens total
  const long tb[3] = {0, 32L * 1024, 32L * 1024 + 32L * 768};

  _Float16* Qw = (_Float16*)d_ws;       // f16 [NT][512]
  _Float16* Kw = Qw + NT * D_MODEL;
  _Float16* Vw = Kw + NT * D_MODEL;
  float* out = (float*)d_out;

  // 1) projections: q/k/v -> ws (f16); o-proj (+bias) -> out (f32)
  for (int m = 0; m < 3; ++m) {
    const int Tm = 32 * Kmods[m];
    dim3 grid((Tm / 64) * 8, 4);
    proj_kernel<<<grid, 256, 0, stream>>>(
        E[m], Wq, Wk, Wv, Wo, bq, bk, bv, bo, Qw + tb[m] * D_MODEL,
        Kw + tb[m] * D_MODEL, Vw + tb[m] * D_MODEL, out + tb[m] * D_MODEL);
  }

  // 2) six cross-attention passes, each accumulating msg into out
  for (int m = 0; m < 3; ++m)
    for (int n = 0; n < 3; ++n) {
      if (n == m) continue;
      const int Km = Kmods[m], Kn = Kmods[n];
      const size_t smem = (size_t)16 * 520 * 2 +        // Qs
                          (size_t)16 * Kn * 4 +         // S
                          (size_t)16 * (Kn + 8) * 2 +   // P
                          (size_t)512 * 40 * 2;         // Vt
      dim3 grid(Km / 16, 32);
      attn_kernel<<<grid, 256, smem, stream>>>(Qw, Kw, Vw, out, Km, Kn, tb[m],
                                               tb[n], tb[m] * D_MODEL);
    }

  // 3) relu in place
  relu_kernel<<<2048, 256, 0, stream>>>(out, NT * (long)D_MODEL);
}
